// MultiHeadAttention_54692113548012
// MI455X (gfx1250) — matrix operand order repaired and checked by reading, not compiled, not run on hardware
//
#include <hip/hip_runtime.h>

// MI455X / gfx1250 multi-head attention, bf16 WMMA + TDM-staged weight tiles.
// B=4, S=1024, D=1024, H=16, d=64.

typedef __attribute__((ext_vector_type(16))) __bf16 v16bf;
typedef __attribute__((ext_vector_type(8)))  __bf16 v8bf;
typedef __attribute__((ext_vector_type(2)))  __bf16 v2bf;
typedef __attribute__((ext_vector_type(8)))  float  v8f;
typedef __attribute__((ext_vector_type(4)))  unsigned int v4u;
typedef __attribute__((ext_vector_type(8)))  int v8i;
typedef __attribute__((ext_vector_type(4)))  int v4i;

#define S_DIM 1024
#define D_DIM 1024
#define KDIM  1024
#define BATCH 4
#define HEADS 16
#define HDIM  64
#define MROWS (BATCH * S_DIM)   // 4096

__device__ __forceinline__ v8f wmma_bf16(v16bf a, v16bf b, v8f c) {
  // (neg_a, A, neg_b, B, c_mod, C, reuse_a, reuse_b)
  return __builtin_amdgcn_wmma_f32_16x16x32_bf16(false, a, false, b, (short)0, c,
                                                 false, false);
}

// A-matrix 16x32 bf16 fragment: lane half h holds contiguous k = 16h..16h+15.
__device__ __forceinline__ v16bf load_a_frag(const __bf16* rowbase, int aklo) {
  union { v16bf v; v8bf h[2]; } u;
  u.h[0] = *(const v8bf*)(rowbase + 2 * aklo);
  u.h[1] = *(const v8bf*)(rowbase + 2 * aklo + 8);
  return u.v;
}

// TDM: stage a [rows x KDIM] bf16 tile (row-major, stride KDIM) into LDS.
// D# layout per CDNA5 ISA ch.8 (group0: count/lds/global/type, group1: dims).
__device__ __forceinline__ void tdm_load_tile_b16(const void* gptr, void* ldsptr,
                                                  unsigned rows) {
  unsigned long long ga = (unsigned long long)(uintptr_t)gptr;
  unsigned la = (unsigned)(uintptr_t)ldsptr;   // low 32 bits = LDS offset
  v4u g0;
  g0[0] = 1u;                                          // count=1, user D#
  g0[1] = la;                                          // lds_addr [63:32]
  g0[2] = (unsigned)ga;                                // global_addr [95:64]
  g0[3] = (unsigned)((ga >> 32) & 0x01FFFFFFu) | (2u << 30);  // ga[56:32] | type=2
  v8i g1;
  g1[0] = (int)(1u << 16);                 // data_size=1 (2 bytes/elem)
  g1[1] = (int)((unsigned)KDIM << 16);     // tensor_dim0[15:0] = 1024
  g1[2] = (int)(rows << 16);               // tensor_dim1 = rows
  g1[3] = (int)((unsigned)KDIM << 16);     // tile_dim0 = 1024
  g1[4] = (int)rows;                       // tile_dim1 = rows (tile_dim2=0)
  g1[5] = (int)KDIM;                       // tensor_dim0_stride = 1024
  g1[6] = 0;                               // stride hi / tensor_dim1_stride lo
  g1[7] = 0;
  v4i g2 = {0, 0, 0, 0};
  v4i g3 = {0, 0, 0, 0};
#if defined(__clang_major__) && __clang_major__ >= 23
  v8i gpad = {0, 0, 0, 0, 0, 0, 0, 0};
  __builtin_amdgcn_tensor_load_to_lds(g0, g1, g2, g3, gpad, 0);
#else
  __builtin_amdgcn_tensor_load_to_lds(g0, g1, g2, g3, 0);
#endif
}

__global__ void cvt_f32_bf16(const float* __restrict__ in, __bf16* __restrict__ out,
                             int n) {
  int i = blockIdx.x * blockDim.x + threadIdx.x;
  if (i < n) out[i] = (__bf16)in[i];
}

// C[m,n] = sum_k A[m,k] * Bw[n,k]   (NT GEMM), M=4096, N=K=1024.
// Block = 8 waves; block owns one 16-col N-tile (weight rows tn*16..+15) staged
// once into LDS by the Tensor Data Mover; each wave computes two 16x16 M-tiles.
// mode 0: bf16 split-head store      obf[b,h,s,dd]   (Q, K projections)
// mode 1: bf16 split-head transposed obf[b,h,dd,s]   (V projection)
// mode 2: f32 plain store            of[m,n]         (output projection)
__global__ void gemm_nt(const __bf16* __restrict__ A, const __bf16* __restrict__ Bw,
                        __bf16* __restrict__ obf, float* __restrict__ of, int mode) {
  __shared__ __bf16 ldsB[16 * KDIM];             // 32 KB weight tile

  const int tn    = blockIdx.x & 63;             // 64 N-tiles
  const int tmblk = blockIdx.x >> 6;             // 16 M-blocks of 16 M-tiles
  const int lane  = threadIdx.x & 31;
  const int wave  = threadIdx.x >> 5;
  const int hf    = lane >> 4;
  const int l16   = lane & 15;

  if (threadIdx.x < 32) {                        // wave 0 issues the DMA
    tdm_load_tile_b16(Bw + (size_t)tn * 16 * KDIM, ldsB, 16u);
    __builtin_amdgcn_s_wait_tensorcnt(0);
  }
  __syncthreads();

  const int aklo   = hf ? 8 : 0;
  const int kshift = hf ? 16 : 0;
  const int tm0 = tmblk * 16 + wave * 2;
  const int tm1 = tm0 + 1;
  const __bf16* arow0 = A + (size_t)(tm0 * 16 + l16) * KDIM;
  const __bf16* arow1 = A + (size_t)(tm1 * 16 + l16) * KDIM;
  const __bf16* brow  = ldsB + (size_t)l16 * KDIM + kshift;

  v8f c0 = {}, c1 = {};
  for (int k0 = 0; k0 < KDIM; k0 += 32) {
    __builtin_prefetch(arow0 + k0 + 256, 0, 1);  // global_prefetch_b8
    v16bf b  = *(const v16bf*)(brow + k0);       // ds_load from staged tile
    v16bf a0 = load_a_frag(arow0 + k0, aklo);
    v16bf a1 = load_a_frag(arow1 + k0, aklo);
    c0 = wmma_bf16(a0, b, c0);
    c1 = wmma_bf16(a1, b, c1);
  }

#pragma unroll
  for (int t = 0; t < 2; ++t) {
    const v8f* cp = t ? &c1 : &c0;
    const int tm = t ? tm1 : tm0;
#pragma unroll
    for (int r = 0; r < 8; ++r) {
      const int m = tm * 16 + r + 8 * hf;        // C/D layout: VGPR r -> M=r(+8)
      const int n = tn * 16 + l16;               // N = lane%16
      const float val = (*cp)[r];
      if (mode == 2) { of[(size_t)m * D_DIM + n] = val; continue; }
      const int b_ = m >> 10, s = m & 1023;
      const int h = n >> 6, dd = n & 63;
      if (mode == 0)
        obf[(((size_t)(b_ * HEADS + h) * S_DIM) + s) * HDIM + dd] = (__bf16)val;
      else
        obf[(((size_t)(b_ * HEADS + h) * HDIM) + dd) * S_DIM + s] = (__bf16)val;
    }
  }
}

// One workgroup (4 waves, 128 threads) per (b, h, 16-row q-tile).
// dynamic LDS: float sc[16][1024] (raw scores); static: bf16 P[16][1024].
__global__ void attn_kernel(const __bf16* __restrict__ qh, const __bf16* __restrict__ kh,
                            const __bf16* __restrict__ vt, const int* __restrict__ vlens,
                            __bf16* __restrict__ ctx) {
  extern __shared__ float sc[];
  __shared__ __bf16 pbf[16 * S_DIM];           // 32 KB packed probabilities
  __shared__ float red[16][8];

  const int blk = blockIdx.x;                  // b*1024 + h*64 + qt
  const int qt  = blk & 63;
  const int h   = (blk >> 6) & 15;
  const int b_  = blk >> 10;
  const int tid  = threadIdx.x;
  const int lane = tid & 31;
  const int wave = tid >> 5;
  const int hf   = lane >> 4;
  const int l16  = lane & 15;
  const int vl   = vlens[b_];

  const __bf16* Q = qh + (size_t)(b_ * HEADS + h) * S_DIM * HDIM;  // [S][64]
  const __bf16* K = kh + (size_t)(b_ * HEADS + h) * S_DIM * HDIM;  // [S][64]
  const __bf16* V = vt + (size_t)(b_ * HEADS + h) * HDIM * S_DIM;  // [64][S]

  const int aklo   = hf ? 8 : 0;
  const int kshift = hf ? 16 : 0;

  const __bf16* qrow = Q + (size_t)(qt * 16 + l16) * HDIM;
  v16bf aq0 = load_a_frag(qrow, aklo);
  v16bf aq1 = load_a_frag(qrow + 32, aklo);

  // ---- scores = (Q Kt) / sqrt(d), masked; each wave handles 16 key-tiles ----
  for (int kt = wave; kt < S_DIM / 16; kt += 4) {
    const int key = kt * 16 + l16;             // B-fragment column = key index
    const __bf16* krow = K + (size_t)key * HDIM + kshift;
    v16bf b0 = *(const v16bf*)(krow);
    v16bf b1 = *(const v16bf*)(krow + 32);
    v8f c = {};
    c = wmma_bf16(aq0, b0, c);
    c = wmma_bf16(aq1, b1, c);
    const bool valid = key < vl;               // per-batch key mask
#pragma unroll
    for (int r = 0; r < 8; ++r) {
      float s = valid ? c[r] * 0.125f : -1.0e6f;   // 1/sqrt(64), MASK_VAL
      sc[(r + 8 * hf) * S_DIM + kt * 16 + l16] = s;
    }
  }
  __syncthreads();

  // ---- softmax over 1024 keys, 8 threads per row; emit packed bf16 P ----
  {
    const int row = tid >> 3, sub = tid & 7;
    float m = -3.0e38f;
    for (int j = sub; j < S_DIM; j += 8) m = fmaxf(m, sc[row * S_DIM + j]);
    red[row][sub] = m;
    __syncthreads();
    float m8 = red[row][0];
#pragma unroll
    for (int j = 1; j < 8; ++j) m8 = fmaxf(m8, red[row][j]);
    float s = 0.f;
    for (int j = sub; j < S_DIM; j += 8) s += __expf(sc[row * S_DIM + j] - m8);
    __syncthreads();
    red[row][sub] = s;
    __syncthreads();
    float s8 = 0.f;
#pragma unroll
    for (int j = 0; j < 8; ++j) s8 += red[row][j];
    const float inv = 1.0f / s8;
    for (int j = sub * 2; j < S_DIM; j += 16) {   // bf16 pair -> one b32 store
      v2bf p;
      p[0] = (__bf16)(__expf(sc[row * S_DIM + j]     - m8) * inv);
      p[1] = (__bf16)(__expf(sc[row * S_DIM + j + 1] - m8) * inv);
      *(v2bf*)(&pbf[row * S_DIM + j]) = p;
    }
  }
  __syncthreads();

  // ---- out(16x64) = P(16x1024) x V(1024x64); wave w -> d-columns [16w,16w+16) ----
  const int coln = wave * 16 + l16;
  const __bf16* vrow = V + (size_t)coln * S_DIM + kshift;   // contiguous in key
  const __bf16* prow = &pbf[l16 * S_DIM];
  v8f c = {};
  for (int k0 = 0; k0 < S_DIM; k0 += 32) {
    v16bf a = load_a_frag(prow + k0, aklo);      // 2x ds_load_b128
    v16bf b = *(const v16bf*)(vrow + k0);
    c = wmma_bf16(a, b, c);
  }
#pragma unroll
  for (int r = 0; r < 8; ++r) {
    const int s = qt * 16 + r + 8 * hf;
    ctx[((size_t)b_ * S_DIM + s) * D_DIM + h * HDIM + coln] = (__bf16)c[r];
  }
}

extern "C" void kernel_launch(void* const* d_in, const int* in_sizes, int n_in,
                              void* d_out, int out_size, void* d_ws, size_t ws_size,
                              hipStream_t stream) {
  (void)in_sizes; (void)n_in; (void)out_size; (void)ws_size;
  const float* q  = (const float*)d_in[0];
  const float* k  = (const float*)d_in[1];
  const float* v  = (const float*)d_in[2];
  const int*   vl = (const int*)d_in[3];
  const float* wq = (const float*)d_in[4];
  const float* wk = (const float*)d_in[5];
  const float* wv = (const float*)d_in[6];
  const float* wo = (const float*)d_in[7];
  float* out = (float*)d_out;

  const size_t E  = (size_t)MROWS * D_DIM;   // 4M activation elems
  const size_t Wn = (size_t)D_DIM * D_DIM;   // 1M weight elems
  __bf16* ws  = (__bf16*)d_ws;               // total: 7E + 4Wn bf16 = 64 MB
  __bf16* xq  = ws;
  __bf16* xk  = ws + E;
  __bf16* xv  = ws + 2 * E;
  __bf16* bwq = ws + 3 * E;
  __bf16* bwk = bwq + Wn;
  __bf16* bwv = bwk + Wn;
  __bf16* bwo = bwv + Wn;
  __bf16* qh  = bwo + Wn;   // [B,H,S,d]
  __bf16* kh  = qh + E;     // [B,H,S,d]
  __bf16* vt  = kh + E;     // [B,H,d,S]
  __bf16* ctx = vt + E;     // [B,S,D]

  const int cvtThreads = 256;
  cvt_f32_bf16<<<(int)((E + 255) / 256), cvtThreads, 0, stream>>>(q, xq, (int)E);
  cvt_f32_bf16<<<(int)((E + 255) / 256), cvtThreads, 0, stream>>>(k, xk, (int)E);
  cvt_f32_bf16<<<(int)((E + 255) / 256), cvtThreads, 0, stream>>>(v, xv, (int)E);
  cvt_f32_bf16<<<(int)((Wn + 255) / 256), cvtThreads, 0, stream>>>(wq, bwq, (int)Wn);
  cvt_f32_bf16<<<(int)((Wn + 255) / 256), cvtThreads, 0, stream>>>(wk, bwk, (int)Wn);
  cvt_f32_bf16<<<(int)((Wn + 255) / 256), cvtThreads, 0, stream>>>(wv, bwv, (int)Wn);
  cvt_f32_bf16<<<(int)((Wn + 255) / 256), cvtThreads, 0, stream>>>(wo, bwo, (int)Wn);

  // 1024 blocks: 64 N-tiles x 16 M-blocks; 8 waves, 2 M-tiles each.
  const int gemmBlocks = 64 * 16;
  gemm_nt<<<gemmBlocks, 256, 0, stream>>>(xq, bwq, qh, nullptr, 0);
  gemm_nt<<<gemmBlocks, 256, 0, stream>>>(xk, bwk, kh, nullptr, 0);
  gemm_nt<<<gemmBlocks, 256, 0, stream>>>(xv, bwv, vt, nullptr, 1);

  attn_kernel<<<BATCH * HEADS * (S_DIM / 16), 128, 16 * S_DIM * sizeof(float),
                stream>>>(qh, kh, vt, vl, ctx);

  gemm_nt<<<gemmBlocks, 256, 0, stream>>>(ctx, bwo, nullptr, out, 2);
}
